// ADEFNet_317827579916
// MI455X (gfx1250) — matrix operand order repaired and checked by reading, not compiled, not run on hardware
//
#include <hip/hip_runtime.h>
#include <hip/hip_bf16.h>

// ---- problem constants (match reference) ----
#define BATCH 16
#define CIN   64
#define HH    64
#define WW    64
#define HP    66              // padded H
#define WPD   66              // padded W
#define NTAP  9
#define OUTC  48
#define OFFC  18              // 2*NTAP offset channels
#define KTOT  576             // CIN * NTAP (GEMM K for both convs)

typedef __attribute__((ext_vector_type(16))) __bf16 v16bf;
typedef __attribute__((ext_vector_type(8)))  __bf16 v8bf;
typedef __attribute__((ext_vector_type(8)))  float  v8f;

__device__ __forceinline__ __bf16 f2bf(float f) {
  unsigned u = __builtin_bit_cast(unsigned, f);
  u += 0x7fffu + ((u >> 16) & 1u);               // round-to-nearest-even
  unsigned short s = (unsigned short)(u >> 16);
  return __builtin_bit_cast(__bf16, s);
}
__device__ __forceinline__ float bf2f(__bf16 h) {
  unsigned short s = __builtin_bit_cast(unsigned short, h);
  unsigned u = ((unsigned)s) << 16;
  return __builtin_bit_cast(float, u);
}
__device__ __forceinline__ int iclamp(int v, int lo, int hi) {
  return v < lo ? lo : (v > hi ? hi : v);
}

// ---------------------------------------------------------------------------
// Kernel P0a: zero the 1-pixel border of the padded NHWC buffer.
// 260 border pixels per image * 64 ch * 16 images = 266,240 elements.
// ---------------------------------------------------------------------------
__global__ void __launch_bounds__(256)
adef_zero_border(unsigned short* __restrict__ xpad_u) {
  __bf16* xpad = reinterpret_cast<__bf16*>(xpad_u);
  int i = blockIdx.x * 256 + threadIdx.x;
  const int NB = BATCH * 260 * CIN;
  if (i >= NB) return;
  int c  = i & 63;
  int t  = i >> 6;
  int pb = t % 260;
  int b  = t / 260;
  int hp, wp;
  if (pb < 66)       { hp = 0;  wp = pb; }
  else if (pb < 132) { hp = 65; wp = pb - 66; }
  else { int r = pb - 132; hp = (r >> 1) + 1; wp = (r & 1) ? 65 : 0; }
  xpad[(((size_t)b * HP + hp) * WPD + wp) * CIN + c] =
      __builtin_bit_cast(__bf16, (unsigned short)0);
}

// ---------------------------------------------------------------------------
// Kernel P0b: x NCHW f32 -> interior of padded NHWC bf16 via LDS transpose.
// Tile: 64 channels x 32 w at fixed (b,h).  Reads coalesced along w
// (128B runs), writes coalesced along c (128B bf16 runs).  tile stride 33
// keeps both LDS phases bank-conflict-free (33 mod 32 == 1).
// ---------------------------------------------------------------------------
__global__ void __launch_bounds__(256)
adef_pad_nhwc(const float* __restrict__ x, unsigned short* __restrict__ xpad_u) {
  __bf16* xpad = reinterpret_cast<__bf16*>(xpad_u);
  __shared__ float tile[CIN * 33];
  int blk = blockIdx.x;              // b*128 + h*2 + wt
  int wt = blk & 1;
  int h  = (blk >> 1) & 63;
  int b  = blk >> 7;
  int w0 = wt * 32;
  int t  = threadIdx.x;
#pragma unroll
  for (int it = 0; it < 8; ++it) {   // load 64c x 32w
    int idx = it * 256 + t;
    int c = idx >> 5;
    int w = idx & 31;
    tile[c * 33 + w] = x[(((size_t)b * CIN + c) * HH + h) * WW + (w0 + w)];
  }
  __syncthreads();
  __bf16* dst = xpad + (((size_t)b * HP + (h + 1)) * WPD + (w0 + 1)) * CIN;
#pragma unroll
  for (int it = 0; it < 8; ++it) {   // store 32w x 64c, c fastest
    int idx = it * 256 + t;
    int c = idx & 63;
    int w = idx >> 6;
    dst[(size_t)w * CIN + c] = f2bf(tile[c * 33 + w]);
  }
}

// ---------------------------------------------------------------------------
// Kernel P1: pack weights to bf16 [o][k] with k = tap*64 + c.
// wpT: 32 rows (18 real + zero pad) x 576.  wcT: 48 x 576.
// Row-major [o][k] => each WMMA B fragment (16 bf16, k-contiguous) is one
// aligned 32B load.
// ---------------------------------------------------------------------------
__global__ void adef_pack_w(const float* __restrict__ w_p,
                            const float* __restrict__ w_c,
                            unsigned short* __restrict__ wpT_u,
                            unsigned short* __restrict__ wcT_u) {
  __bf16* wpT = reinterpret_cast<__bf16*>(wpT_u);
  __bf16* wcT = reinterpret_cast<__bf16*>(wcT_u);
  int i = blockIdx.x * blockDim.x + threadIdx.x;
  if (i < 32 * KTOT) {
    int k = i % KTOT, o = i / KTOT;
    int c = k & 63, tap = k >> 6;
    float v = 0.0f;
    if (o < OFFC) v = w_p[(((size_t)o * CIN + c) * 3 + tap / 3) * 3 + (tap % 3)];
    wpT[i] = f2bf(v);
  } else if (i < (32 + OUTC) * KTOT) {
    int j = i - 32 * KTOT;
    int k = j % KTOT, o = j / KTOT;
    int c = k & 63, tap = k >> 6;
    wcT[j] = f2bf(w_c[(((size_t)o * CIN + c) * 3 + tap / 3) * 3 + (tap % 3)]);
  }
}

// ---------------------------------------------------------------------------
// Kernel K1: offset conv (3x3, 64->18, pad 1) as implicit GEMM with
// v_wmma_f32_16x16x32_bf16.  M = 65536 pixels (wave owns 16 consecutive w),
// K = 576, N = 32 (18 real).  Writes offsets NHWC f32 [pix][18] (+bias).
// ---------------------------------------------------------------------------
__global__ void __launch_bounds__(128)
adef_offset_conv(const unsigned short* __restrict__ xpad_u,
                 const unsigned short* __restrict__ wpT_u,
                 const float* __restrict__ b_p,
                 float* __restrict__ offs) {
  const __bf16* xpad = reinterpret_cast<const __bf16*>(xpad_u);
  const __bf16* wpT  = reinterpret_cast<const __bf16*>(wpT_u);
  int lane = threadIdx.x & 31;
  int wave = threadIdx.x >> 5;
  int tile = blockIdx.x * 4 + wave;     // 4096 tiles total, exact cover
  int m0   = tile * 16;
  int mrow = lane & 15;                 // A row / B column
  int hi   = lane >> 4;                 // K-half selector (contiguous order k = 16*hi + e)
  int bb = m0 >> 12;
  int h0 = (m0 >> 6) & 63;
  int wl = (m0 & 63) + mrow;            // this lane's pixel w

  v8f acc0 = {};                        // out channels 0..15
  v8f acc1 = {};                        // out channels 16..31 (16,17 real)
#pragma unroll 1
  for (int tap = 0; tap < NTAP; ++tap) {
    int kh = tap / 3, kw = tap % 3;
    const __bf16* srow =
        xpad + (((size_t)bb * HP + (h0 + kh)) * WPD + (wl + kw)) * CIN;
#pragma unroll
    for (int ch = 0; ch < 2; ++ch) {
      int chalf = ch * 32;
      // A fragment: elems 0..15 -> c = chalf+hi*16+e
      v8bf r0 = *reinterpret_cast<const v8bf*>(srow + chalf + hi * 16);
      v8bf r1 = *reinterpret_cast<const v8bf*>(srow + chalf + hi * 16 + 8);
      v16bf a = __builtin_shufflevector(r0, r1, 0, 1, 2, 3, 4, 5, 6, 7,
                                        8, 9, 10, 11, 12, 13, 14, 15);
      int kk = tap * 64 + chalf + hi * 16;
      v16bf wb0 = *reinterpret_cast<const v16bf*>(wpT + (size_t)mrow * KTOT + kk);
      v16bf wb1 = *reinterpret_cast<const v16bf*>(wpT + (size_t)(16 + mrow) * KTOT + kk);
      acc0 = __builtin_amdgcn_wmma_f32_16x16x32_bf16(false, a, false, wb0,
                                                     (short)0, acc0, false, false);
      acc1 = __builtin_amdgcn_wmma_f32_16x16x32_bf16(false, a, false, wb1,
                                                     (short)0, acc1, false, false);
    }
  }
  // D layout: col = mrow, rows r + hi*8
  float bia0 = b_p[mrow];
#pragma unroll
  for (int r = 0; r < 8; ++r) {
    size_t pix = (size_t)(m0 + r + hi * 8);
    offs[pix * OFFC + mrow] = acc0[r] + bia0;
  }
  if (mrow < 2) {
    float bia1 = b_p[16 + mrow];
#pragma unroll
    for (int r = 0; r < 8; ++r) {
      size_t pix = (size_t)(m0 + r + hi * 8);
      offs[pix * OFFC + 16 + mrow] = acc1[r] + bia1;
    }
  }
}

// ---------------------------------------------------------------------------
// Kernel K2: fused bilinear deformable sampling + final conv.
// out[b,o,h,w] = sum_{c,tap} w_c[o,c,tap] * bilin(xpad; p0+pn+off)[c,tap]
// GEMM: M = 65536, K = 576 (tap-major), N = 48 (3 WMMA N-tiles).
// Per-pixel offsets hoisted to registers (one load clause); tap loop fully
// unrolled so they stay in registers and gathers pipeline across taps.
// ---------------------------------------------------------------------------
__global__ void __launch_bounds__(128)
adef_deform_conv(const unsigned short* __restrict__ xpad_u,
                 const unsigned short* __restrict__ wcT_u,
                 const float* __restrict__ offs,
                 float* __restrict__ out) {
  const __bf16* xpad = reinterpret_cast<const __bf16*>(xpad_u);
  const __bf16* wcT  = reinterpret_cast<const __bf16*>(wcT_u);
  int lane = threadIdx.x & 31;
  int wave = threadIdx.x >> 5;
  int tile = blockIdx.x * 4 + wave;
  int m0   = tile * 16;
  int mrow = lane & 15;
  int hi   = lane >> 4;
  int bb = m0 >> 12;
  int h0 = (m0 >> 6) & 63;
  int wl = (m0 & 63) + mrow;
  size_t pix = (size_t)(m0 + mrow);

  // hoist all 18 per-pixel offsets (72 contiguous bytes) into registers
  float offv[OFFC];
#pragma unroll
  for (int i = 0; i < OFFC; ++i) offv[i] = offs[pix * OFFC + i];

  v8f acc0 = {}, acc1 = {}, acc2 = {};
#pragma unroll
  for (int tap = 0; tap < NTAP; ++tap) {
    // sampling position p = p0 + p_n + offset (coords live in padded space)
    float offx = offv[tap];
    float offy = offv[NTAP + tap];
    float px = (float)(h0 + (tap / 3)) + offx;     // (h0+1) + (tap/3 - 1)
    float py = (float)(wl + (tap % 3)) + offy;     // (wl+1) + (tap%3 - 1)
    float flx = floorf(px), fly = floorf(py);
    int qltx = iclamp((int)flx, 0, HP - 1);
    int qlty = iclamp((int)fly, 0, WPD - 1);
    int qrbx = iclamp((int)flx + 1, 0, HP - 1);
    int qrby = iclamp((int)fly + 1, 0, WPD - 1);
    float pxc = fminf(fmaxf(px, 0.0f), (float)(HP - 1));
    float pyc = fminf(fmaxf(py, 0.0f), (float)(WPD - 1));
    float gx_l = 1.0f + ((float)qltx - pxc);
    float gx_r = 1.0f - ((float)qrbx - pxc);
    float gy_l = 1.0f + ((float)qlty - pyc);
    float gy_r = 1.0f - ((float)qrby - pyc);
    float g_lt = gx_l * gy_l, g_rb = gx_r * gy_r;
    float g_lb = gx_l * gy_r, g_rt = gx_r * gy_l;
    const __bf16* plt = xpad + (((size_t)bb * HP + qltx) * WPD + qlty) * CIN;
    const __bf16* prb = xpad + (((size_t)bb * HP + qrbx) * WPD + qrby) * CIN;
    const __bf16* plb = xpad + (((size_t)bb * HP + qltx) * WPD + qrby) * CIN;
    const __bf16* prt = xpad + (((size_t)bb * HP + qrbx) * WPD + qlty) * CIN;
#pragma unroll
    for (int chv = 0; chv < 2; ++chv) {
      int c0 = chv * 32 + hi * 16;
      int c1 = chv * 32 + hi * 16 + 8;
      v8bf lt0 = *reinterpret_cast<const v8bf*>(plt + c0);
      v8bf lt1 = *reinterpret_cast<const v8bf*>(plt + c1);
      v8bf rb0 = *reinterpret_cast<const v8bf*>(prb + c0);
      v8bf rb1 = *reinterpret_cast<const v8bf*>(prb + c1);
      v8bf lb0 = *reinterpret_cast<const v8bf*>(plb + c0);
      v8bf lb1 = *reinterpret_cast<const v8bf*>(plb + c1);
      v8bf rt0 = *reinterpret_cast<const v8bf*>(prt + c0);
      v8bf rt1 = *reinterpret_cast<const v8bf*>(prt + c1);
      v16bf a;
#pragma unroll
      for (int e = 0; e < 8; ++e) {
        float v0 = g_lt * bf2f(lt0[e]) + g_rb * bf2f(rb0[e]) +
                   g_lb * bf2f(lb0[e]) + g_rt * bf2f(rt0[e]);
        float v1 = g_lt * bf2f(lt1[e]) + g_rb * bf2f(rb1[e]) +
                   g_lb * bf2f(lb1[e]) + g_rt * bf2f(rt1[e]);
        a[e]     = f2bf(v0);
        a[e + 8] = f2bf(v1);
      }
      int kk = tap * 64 + chv * 32 + hi * 16;
      v16bf w0 = *reinterpret_cast<const v16bf*>(wcT + (size_t)(0  + mrow) * KTOT + kk);
      v16bf w1 = *reinterpret_cast<const v16bf*>(wcT + (size_t)(16 + mrow) * KTOT + kk);
      v16bf w2 = *reinterpret_cast<const v16bf*>(wcT + (size_t)(32 + mrow) * KTOT + kk);
      acc0 = __builtin_amdgcn_wmma_f32_16x16x32_bf16(false, a, false, w0,
                                                     (short)0, acc0, false, false);
      acc1 = __builtin_amdgcn_wmma_f32_16x16x32_bf16(false, a, false, w1,
                                                     (short)0, acc1, false, false);
      acc2 = __builtin_amdgcn_wmma_f32_16x16x32_bf16(false, a, false, w2,
                                                     (short)0, acc2, false, false);
    }
  }
  // Store NCHW f32: o = nt*16 + mrow, rows m = r + hi*8 -> w = (m0&63)+m
#pragma unroll
  for (int r = 0; r < 8; ++r) {
    int m = r + hi * 8;
    int wout = (m0 & 63) + m;
    size_t base = (((size_t)bb * OUTC) * HH + h0) * WW + wout;
    out[base + (size_t)(0  + mrow) * HH * WW] = acc0[r];
    out[base + (size_t)(16 + mrow) * HH * WW] = acc1[r];
    out[base + (size_t)(32 + mrow) * HH * WW] = acc2[r];
  }
}

// ---------------------------------------------------------------------------
extern "C" void kernel_launch(void* const* d_in, const int* in_sizes, int n_in,
                              void* d_out, int out_size, void* d_ws, size_t ws_size,
                              hipStream_t stream) {
  const float* x   = (const float*)d_in[0];
  const float* w_p = (const float*)d_in[1];
  const float* b_p = (const float*)d_in[2];
  const float* w_c = (const float*)d_in[3];
  float* out = (float*)d_out;

  // workspace layout (all 256B aligned)
  const size_t XPAD_ELEMS = (size_t)BATCH * HP * WPD * CIN;      // 4,460,544
  const size_t OFFS_ELEMS = (size_t)BATCH * HH * WW * OFFC;      // 1,179,648
  char* ws = (char*)d_ws;
  size_t o0 = 0;                                   // xpad bf16
  size_t o1 = o0 + ((XPAD_ELEMS * 2 + 255) & ~(size_t)255);  // offs f32
  size_t o2 = o1 + ((OFFS_ELEMS * 4 + 255) & ~(size_t)255);  // wpT bf16 32x576
  size_t o3 = o2 + ((32 * KTOT * 2 + 255) & ~(size_t)255);   // wcT bf16 48x576
  unsigned short* xpad_u = (unsigned short*)(ws + o0);
  float*          offs   = (float*)(ws + o1);
  unsigned short* wpT_u  = (unsigned short*)(ws + o2);
  unsigned short* wcT_u  = (unsigned short*)(ws + o3);

  // border zero + interior LDS transpose (disjoint writes, same stream)
  adef_zero_border<<<(BATCH * 260 * CIN + 255) / 256, 256, 0, stream>>>(xpad_u);
  adef_pad_nhwc<<<BATCH * HH * (WW / 32), 256, 0, stream>>>(x, xpad_u);
  adef_pack_w<<<((32 + OUTC) * KTOT + 255) / 256, 256, 0, stream>>>(w_p, w_c, wpT_u, wcT_u);
  // 65536 pixels / 16 per wave = 4096 tiles; 4 waves (128 thr) per block
  adef_offset_conv<<<1024, 128, 0, stream>>>(xpad_u, wpT_u, b_p, offs);
  adef_deform_conv<<<1024, 128, 0, stream>>>(xpad_u, wcT_u, offs, out);
}
